// HSLSTMRegressor_49065706389927
// MI455X (gfx1250) — matrix operand order repaired and checked by reading, not compiled, not run on hardware
//
#include <hip/hip_runtime.h>
#include <stdint.h>

#define B_SZ 128
#define T_SZ 256
#define IN_SZ 64
#define H_SZ 512
#define L_SZ 3
#define G4 (4 * H_SZ) // 2048 gate rows

typedef __attribute__((ext_vector_type(16))) __bf16   v16bf;
typedef __attribute__((ext_vector_type(8)))  float    v8f;
typedef __attribute__((ext_vector_type(4)))  uint32_t u32x4;

union FragCast { v16bf v; u32x4 u[2]; };

__device__ __forceinline__ uint16_t f2bf(float x) {
  uint32_t u = __float_as_uint(x);
  uint32_t r = u + 0x7FFFu + ((u >> 16) & 1u); // round-to-nearest-even
  return (uint16_t)(r >> 16);
}
__device__ __forceinline__ float bf2f(uint16_t h) {
  return __uint_as_float(((uint32_t)h) << 16);
}
__device__ __forceinline__ float fast_sigmoid(float x) {
  return 1.0f / (1.0f + __expf(-x));
}
__device__ __forceinline__ float fast_tanh(float x) {
  return 2.0f * fast_sigmoid(2.0f * x) - 1.0f;
}

// A fragment (16x32 bf16, row-major source, row = m, contiguous K):
// lanes 0-15 hold K = {k0..k0+7, k0+16..k0+23}; lanes 16-31 the +8 phase.
__device__ __forceinline__ v16bf load_a_frag(const uint16_t* __restrict__ rowPtr,
                                             int k0, int grp) {
  FragCast f;
  const uint16_t* p = rowPtr + k0 + grp * 8;
  f.u[0] = *(const u32x4*)(p);
  f.u[1] = *(const u32x4*)(p + 16);
  return f.v;
}
// B fragment (32x16 bf16). Column n of B = row n of row-major W[N,K].
// lanes 0-15 hold K = {k0..k0+7, k0+16..k0+23}; lanes 16-31 the +8 phase.
__device__ __forceinline__ v16bf load_b_frag(const uint16_t* __restrict__ rowPtr,
                                             int k0, int grp) {
  FragCast f;
  const uint16_t* p = rowPtr + k0 + grp * 8;
  f.u[0] = *(const u32x4*)(p);
  f.u[1] = *(const u32x4*)(p + 16);
  return f.v;
}

__device__ __forceinline__ v8f wmma_bf16(v16bf a, v16bf b, v8f c) {
  return __builtin_amdgcn_wmma_f32_16x16x32_bf16(
      /*neg_a=*/false, a, /*neg_b=*/false, b,
      /*c_mod=*/(short)0, c, /*reuse_a=*/false, /*reuse_b=*/false);
}

// One LSTM cell update for one (t, layer):
//   gates[128,2048] = Ain[128,kIn] @ Wih^T + Hprev[128,512] @ Whh^T + biasSum
//   then fused i/f/g/o pointwise update of c (fp32) and h (bf16).
// Each wave: one 16M x 16N tile replicated over the 4 gate blocks.
__global__ __launch_bounds__(256) void lstm_step_kernel(
    const uint16_t* __restrict__ Ain, int aStride, int kIn,
    const uint16_t* __restrict__ Wih,    // [2048, kIn] bf16
    const uint16_t* __restrict__ Hprev,  // [128, 512]  bf16
    const uint16_t* __restrict__ Whh,    // [2048, 512] bf16
    const float*    __restrict__ biasSum,// [2048]
    float*          __restrict__ Cst,    // [128, 512]  fp32 (in/out)
    uint16_t*       __restrict__ Hout)   // [128, 512]  bf16 (out)
{
  const int tid   = threadIdx.x;
  const int wave  = tid >> 5;
  const int lane  = tid & 31;
  const int grp   = lane >> 4;   // half-wave K phase
  const int lm    = lane & 15;
  const int Wg    = blockIdx.x * 8 + wave;
  const int mTile = Wg & 7;      // 8 M tiles cover B=128
  const int nTile = Wg >> 3;     // 32 N tiles cover H=512 (per gate)
  const int nLoc  = nTile * 16 + lm;   // column within one gate block

  v8f acc0 = {0.f,0.f,0.f,0.f,0.f,0.f,0.f,0.f};
  v8f acc1 = acc0, acc2 = acc0, acc3 = acc0;

  // ---- input projection: A[128,kIn] x Wih^T ----
  {
    const uint16_t* aRow = Ain + (size_t)(mTile * 16 + lm) * aStride;
    const uint16_t* w0 = Wih + (size_t)(0 * H_SZ + nLoc) * kIn;
    const uint16_t* w1 = Wih + (size_t)(1 * H_SZ + nLoc) * kIn;
    const uint16_t* w2 = Wih + (size_t)(2 * H_SZ + nLoc) * kIn;
    const uint16_t* w3 = Wih + (size_t)(3 * H_SZ + nLoc) * kIn;
    for (int k0 = 0; k0 < kIn; k0 += 32) {
      v16bf a = load_a_frag(aRow, k0, grp);
      acc0 = wmma_bf16(a, load_b_frag(w0, k0, grp), acc0);
      acc1 = wmma_bf16(a, load_b_frag(w1, k0, grp), acc1);
      acc2 = wmma_bf16(a, load_b_frag(w2, k0, grp), acc2);
      acc3 = wmma_bf16(a, load_b_frag(w3, k0, grp), acc3);
    }
  }
  // ---- recurrent projection: Hprev[128,512] x Whh^T ----
  {
    const uint16_t* aRow = Hprev + (size_t)(mTile * 16 + lm) * H_SZ;
    const uint16_t* w0 = Whh + (size_t)(0 * H_SZ + nLoc) * H_SZ;
    const uint16_t* w1 = Whh + (size_t)(1 * H_SZ + nLoc) * H_SZ;
    const uint16_t* w2 = Whh + (size_t)(2 * H_SZ + nLoc) * H_SZ;
    const uint16_t* w3 = Whh + (size_t)(3 * H_SZ + nLoc) * H_SZ;
    for (int k0 = 0; k0 < H_SZ; k0 += 32) {
      __builtin_prefetch(w0 + k0 + 64, 0, 0); // global_prefetch_b8 on next chunk
      v16bf a = load_a_frag(aRow, k0, grp);
      acc0 = wmma_bf16(a, load_b_frag(w0, k0, grp), acc0);
      acc1 = wmma_bf16(a, load_b_frag(w1, k0, grp), acc1);
      acc2 = wmma_bf16(a, load_b_frag(w2, k0, grp), acc2);
      acc3 = wmma_bf16(a, load_b_frag(w3, k0, grp), acc3);
    }
  }

  // ---- fused LSTM pointwise epilogue (gates live in this wave's registers) ----
  const float bI = biasSum[0 * H_SZ + nLoc];
  const float bF = biasSum[1 * H_SZ + nLoc];
  const float bG = biasSum[2 * H_SZ + nLoc];
  const float bO = biasSum[3 * H_SZ + nLoc];
  const int rowBase = mTile * 16 + grp * 8; // C layout: VGPR r -> M = rowBase + r
#pragma unroll
  for (int r = 0; r < 8; ++r) {
    const int b = rowBase + r;
    const size_t idx = (size_t)b * H_SZ + nLoc;
    const float ig = fast_sigmoid(acc0[r] + bI);
    const float fg = fast_sigmoid(acc1[r] + bF);
    const float gg = fast_tanh   (acc2[r] + bG);
    const float og = fast_sigmoid(acc3[r] + bO);
    const float cn = fg * Cst[idx] + ig * gg;
    Cst[idx]  = cn;
    Hout[idx] = f2bf(og * fast_tanh(cn));
  }
}

__global__ void f32_to_bf16_kernel(const float* __restrict__ src,
                                   uint16_t* __restrict__ dst, int n) {
  int i = blockIdx.x * blockDim.x + threadIdx.x;
  if (i < n) dst[i] = f2bf(src[i]);
}

__global__ void bias_fuse_kernel(const float* __restrict__ bih0,
                                 const float* __restrict__ bhh0,
                                 const float* __restrict__ bihr,
                                 const float* __restrict__ bhhr,
                                 float* __restrict__ biasSum) { // [3][2048]
  int i = blockIdx.x * blockDim.x + threadIdx.x;
  if (i >= L_SZ * G4) return;
  int l = i / G4, n = i % G4;
  biasSum[i] = (l == 0) ? (bih0[n] + bhh0[n])
                        : (bihr[(size_t)(l - 1) * G4 + n] + bhhr[(size_t)(l - 1) * G4 + n]);
}

// out[b] = h_last[b,:] . Wout + bout ; d_out[128] = mean((out - y)^2)
__global__ __launch_bounds__(128) void output_kernel(
    const uint16_t* __restrict__ hLast, // [128,512] bf16
    const float* __restrict__ Wout,     // [512]
    const float* __restrict__ bout,     // [1]
    const float* __restrict__ y,        // [128,1]
    float* __restrict__ dOut)           // [129]
{
  __shared__ float s[B_SZ];
  const int b = threadIdx.x;
  float acc = 0.f;
  for (int j = 0; j < H_SZ; ++j)
    acc += bf2f(hLast[(size_t)b * H_SZ + j]) * Wout[j];
  const float out = acc + bout[0];
  dOut[b] = out;
  const float d = out - y[b];
  s[b] = d * d;
  __syncthreads();
  for (int st = 64; st > 0; st >>= 1) {
    if (b < st) s[b] += s[b + st];
    __syncthreads();
  }
  if (b == 0) dOut[B_SZ] = s[0] / (float)B_SZ;
}

extern "C" void kernel_launch(void* const* d_in, const int* in_sizes, int n_in,
                              void* d_out, int out_size, void* d_ws, size_t ws_size,
                              hipStream_t stream) {
  (void)in_sizes; (void)n_in; (void)out_size; (void)ws_size;
  const float* x    = (const float*)d_in[0];
  const float* y    = (const float*)d_in[1];
  const float* Wih0 = (const float*)d_in[2];
  const float* Whh0 = (const float*)d_in[3];
  const float* bih0 = (const float*)d_in[4];
  const float* bhh0 = (const float*)d_in[5];
  const float* Wihr = (const float*)d_in[6];
  const float* Whhr = (const float*)d_in[7];
  const float* bihr = (const float*)d_in[8];
  const float* bhhr = (const float*)d_in[9];
  const float* Wout = (const float*)d_in[10];
  const float* bout = (const float*)d_in[11];

  // ---- workspace layout (bytes, all 256B aligned) ----
  constexpr size_t SZ_XBF  = (size_t)B_SZ * T_SZ * IN_SZ * 2;
  constexpr size_t SZ_W0I  = (size_t)G4 * IN_SZ * 2;
  constexpr size_t SZ_W0H  = (size_t)G4 * H_SZ * 2;
  constexpr size_t SZ_WRI  = (size_t)2 * G4 * H_SZ * 2;
  constexpr size_t SZ_WRH  = (size_t)2 * G4 * H_SZ * 2;
  constexpr size_t SZ_BIAS = (size_t)L_SZ * G4 * 4;
  constexpr size_t SZ_H    = (size_t)2 * L_SZ * B_SZ * H_SZ * 2; // ping-pong bf16
  constexpr size_t OFF_XBF  = 0;
  constexpr size_t OFF_W0I  = OFF_XBF + SZ_XBF;
  constexpr size_t OFF_W0H  = OFF_W0I + SZ_W0I;
  constexpr size_t OFF_WRI  = OFF_W0H + SZ_W0H;
  constexpr size_t OFF_WRH  = OFF_WRI + SZ_WRI;
  constexpr size_t OFF_BIAS = OFF_WRH + SZ_WRH;
  constexpr size_t OFF_H    = OFF_BIAS + SZ_BIAS;
  constexpr size_t OFF_C    = OFF_H + SZ_H;
  constexpr size_t SZ_C     = (size_t)L_SZ * B_SZ * H_SZ * 4;

  char* ws = (char*)d_ws;
  uint16_t* xbf     = (uint16_t*)(ws + OFF_XBF);
  uint16_t* w0i     = (uint16_t*)(ws + OFF_W0I);
  uint16_t* w0h     = (uint16_t*)(ws + OFF_W0H);
  uint16_t* wri     = (uint16_t*)(ws + OFF_WRI);
  uint16_t* wrh     = (uint16_t*)(ws + OFF_WRH);
  float*    biasSum = (float*)   (ws + OFF_BIAS);
  uint16_t* hbuf    = (uint16_t*)(ws + OFF_H);
  float*    cstate  = (float*)   (ws + OFF_C);

  auto hptr = [&](int par, int l) {
    return hbuf + ((size_t)par * L_SZ + l) * B_SZ * H_SZ;
  };

  // zero h ping-pong buffers and c state (h(-1)=0, c(-1)=0); graph-capture safe
  hipMemsetAsync(ws + OFF_H, 0, SZ_H + SZ_C, stream);

  // ---- one-time bf16 conversions ----
  int n;
  n = B_SZ * T_SZ * IN_SZ;
  f32_to_bf16_kernel<<<(n + 255) / 256, 256, 0, stream>>>(x, xbf, n);
  n = G4 * IN_SZ;
  f32_to_bf16_kernel<<<(n + 255) / 256, 256, 0, stream>>>(Wih0, w0i, n);
  n = G4 * H_SZ;
  f32_to_bf16_kernel<<<(n + 255) / 256, 256, 0, stream>>>(Whh0, w0h, n);
  n = 2 * G4 * H_SZ;
  f32_to_bf16_kernel<<<(n + 255) / 256, 256, 0, stream>>>(Wihr, wri, n);
  f32_to_bf16_kernel<<<(n + 255) / 256, 256, 0, stream>>>(Whhr, wrh, n);
  n = L_SZ * G4;
  bias_fuse_kernel<<<(n + 255) / 256, 256, 0, stream>>>(bih0, bhh0, bihr, bhhr, biasSum);

  // ---- sequential LSTM: T steps x L layers, ordered by stream ----
  for (int t = 0; t < T_SZ; ++t) {
    const int cur = t & 1, prv = cur ^ 1;
    for (int l = 0; l < L_SZ; ++l) {
      const uint16_t* Ain; int aStride, kIn; const uint16_t* Wih;
      if (l == 0) {
        Ain = xbf + (size_t)t * IN_SZ; aStride = T_SZ * IN_SZ; kIn = IN_SZ; Wih = w0i;
      } else {
        Ain = hptr(cur, l - 1); aStride = H_SZ; kIn = H_SZ;
        Wih = wri + (size_t)(l - 1) * G4 * H_SZ;
      }
      const uint16_t* Hprev = hptr(prv, l);
      const uint16_t* Whh = (l == 0) ? w0h : wrh + (size_t)(l - 1) * G4 * H_SZ;
      lstm_step_kernel<<<dim3(32), dim3(256), 0, stream>>>(
          Ain, aStride, kIn, Wih, Hprev, Whh,
          biasSum + (size_t)l * G4, cstate + (size_t)l * B_SZ * H_SZ,
          hptr(cur, l));
    }
  }

  // ---- final projection + loss ----
  output_kernel<<<1, B_SZ, 0, stream>>>(
      hptr((T_SZ - 1) & 1, L_SZ - 1), Wout, bout, y, (float*)d_out);
}
